// GraphSage_layer_4_56126632624275
// MI455X (gfx1250) — compile-verified, matrix operand order checked by reading, not run on hardware
//
#include <hip/hip_runtime.h>
#include <stdint.h>

// ---------------------------------------------------------------------------
// GraphSAGE (4 layers) on MI455X / gfx1250.
//   S = adj @ h           : bf16 WMMA GEMM, adj converted to bf16 (128MB, fits
//                           192MB L2 -> layers 2..4 stream adj from L2)
//   h' = relu([h|S]W + b) : same WMMA GEMM with concat-A and fused epilogue
// Data staged global->LDS via CDNA5 async copies (ASYNCcnt), double buffered.
// Compute loop batches all fragment ds_loads before the WMMA chain so the
// 8 v_wmma_f32_16x16x32_bf16 issue back-to-back (single s_wait_dscnt).
// ---------------------------------------------------------------------------

typedef __bf16 bf16;
typedef __bf16 v8bf  __attribute__((ext_vector_type(8)));
typedef __bf16 v16bf __attribute__((ext_vector_type(16)));
typedef float  v8f   __attribute__((ext_vector_type(8)));

#define GN     8192
#define GNFEAT 256
#define GNHID  128
#define GNCLS  64

// 16-byte async global->LDS copy (per-lane). dst = LDS byte address (low 32
// bits of a generic pointer into __shared__), src = 64-bit global address.
__device__ __forceinline__ void async_cp16(uint32_t lds_addr, const void* g) {
  asm volatile("global_load_async_to_lds_b128 %0, %1, off"
               :: "v"(lds_addr), "v"((uint64_t)(uintptr_t)g)
               : "memory");
}

// Load a 16x32 bf16 fragment half-pair: lanes 0-15 read K{0..7,16..23} of
// their row, lanes 16-31 read K{8..15,24..31} (per CDNA5 WMMA VGPR layout).
__device__ __forceinline__ v16bf ldfrag(const unsigned char* p) {
  v8bf lo = *(const v8bf*)(p);
  v8bf hi = *(const v8bf*)(p + 32);
  return __builtin_shufflevector(lo, hi, 0,1,2,3,4,5,6,7,8,9,10,11,12,13,14,15);
}

// C[M,N] = A[M,K] @ B[K,N], with B given transposed (Bt[N][K], row stride ldBt).
// CONCAT: A row k<Khalf comes from A0[m*Khalf+k], else A1[m*Khalf+(k-Khalf)].
// EPI: 0 = store bf16 row-major (aggregation output S)
//      1 = bias+ReLU, store bf16 row-major AND transposed (hidden layer)
//      2 = bias, store fp32 row-major (final layer)
template<int BN, bool CONCAT, int EPI>
__global__ __launch_bounds__(256) void gemm_bf16_kernel(
    const bf16* __restrict__ A0, const bf16* __restrict__ A1,
    int ldA, int Khalf,
    const bf16* __restrict__ Bt, int ldBt, int K,
    int M, int Nout, const float* __restrict__ bias,
    bf16* __restrict__ out_row, bf16* __restrict__ out_tr,
    float* __restrict__ out_f)
{
  constexpr int BM    = 128;
  constexpr int BK    = 32;
  constexpr int NT    = BN / 16;            // column tiles per wave
  constexpr int AROWB = BK * 2 + 48;        // 112B row pitch: 16B aligned,
                                            // bank-conflict-free frag reads
  constexpr int A_BUF = BM * AROWB;
  constexpr int B_BUF = BN * AROWB;
  constexpr int AU = (BM * BK * 2 / 16) / 256;   // 16B units per thread (A) = 2
  constexpr int BU = (BN * BK * 2 / 16) / 256;   // (B) = 2 or 1
  constexpr int COPIES = AU + BU;                // async instrs per wave/buffer

  __shared__ __attribute__((aligned(16))) unsigned char smem[2 * A_BUF + 2 * B_BUF];

  const int tid    = threadIdx.x;
  const int wave   = tid >> 5;
  const int lane   = tid & 31;
  const int laneLo = lane & 15;
  const int laneHi = lane >> 4;
  const int blockM = blockIdx.x * BM;
  const int blockN = blockIdx.y * BN;
  const int m0     = wave * 16;              // this wave's 16-row stripe

  v8f acc[NT] = {};

  const uint32_t smem_base = (uint32_t)(uintptr_t)&smem[0];

  auto stage = [&](int buf, int kbase) {
    const bf16* Asrc; int kk, lda;
    if (CONCAT) {
      if (kbase < Khalf) { Asrc = A0; kk = kbase; }
      else               { Asrc = A1; kk = kbase - Khalf; }
      lda = Khalf;
    } else { Asrc = A0; kk = kbase; lda = ldA; }
#pragma unroll
    for (int j = 0; j < AU; ++j) {
      int u = tid + j * 256;
      int r = u >> 2, c = u & 3;                     // 4 x 16B units per row
      const bf16* g = Asrc + (size_t)(blockM + r) * lda + kk + c * 8;
      async_cp16(smem_base + buf * A_BUF + r * AROWB + c * 16, g);
    }
#pragma unroll
    for (int j = 0; j < BU; ++j) {
      int u = tid + j * 256;
      int r = u >> 2, c = u & 3;
      const bf16* g = Bt + (size_t)(blockN + r) * ldBt + kbase + c * 8;
      async_cp16(smem_base + 2 * A_BUF + buf * B_BUF + r * AROWB + c * 16, g);
    }
  };

  const int nk = K / BK;
  stage(0, 0);
  for (int ki = 0; ki < nk; ++ki) {
    const int cur = ki & 1;
    if (ki + 1 < nk) {
      stage(cur ^ 1, (ki + 1) * BK);
      // async ops complete in-order per wave: <=COPIES outstanding means the
      // current buffer's copies have landed.
      asm volatile("s_wait_asynccnt %0" :: "n"(COPIES) : "memory");
    } else {
      asm volatile("s_wait_asynccnt 0" ::: "memory");
    }
    __syncthreads();

    const unsigned char* As = smem + cur * A_BUF;
    const unsigned char* Bs = smem + 2 * A_BUF + cur * B_BUF;

    // Batch all fragment loads, then run the WMMA chain back-to-back.
    v16bf a = ldfrag(As + (m0 + laneLo) * AROWB + laneHi * 16);
    v16bf b[NT];
#pragma unroll
    for (int t = 0; t < NT; ++t)
      b[t] = ldfrag(Bs + (t * 16 + laneLo) * AROWB + laneHi * 16);
#pragma unroll
    for (int t = 0; t < NT; ++t)
      acc[t] = __builtin_amdgcn_wmma_f32_16x16x32_bf16(
          false, a, false, b[t], (short)0, acc[t], false, false);

    __syncthreads();   // all waves done reading before buffer is restaged
  }

  // Epilogue. C layout: VGPR v holds row m0+v (lanes 0-15) / m0+8+v (16-31),
  // column = laneLo within the tile.
#pragma unroll
  for (int t = 0; t < NT; ++t) {
    const int n = blockN + t * 16 + laneLo;
    const float bv = (EPI >= 1) ? bias[n] : 0.0f;
#pragma unroll
    for (int v = 0; v < 8; ++v) {
      const int m = blockM + m0 + laneHi * 8 + v;
      float val = acc[t][v] + bv;
      if (EPI == 1) {
        val = fmaxf(val, 0.0f);
        out_row[(size_t)m * Nout + n] = (bf16)val;
        out_tr[(size_t)n * M + m]     = (bf16)val;
      } else if (EPI == 2) {
        out_f[(size_t)m * Nout + n] = val;
      } else {
        out_row[(size_t)m * Nout + n] = (bf16)val;
      }
    }
  }
}

// fp32 -> bf16, 8 elements per thread (b128 in / b128 out).
__global__ void cvt_f32_bf16_kernel(const float* __restrict__ in,
                                    bf16* __restrict__ out, size_t n) {
  size_t i = ((size_t)blockIdx.x * blockDim.x + threadIdx.x) * 8;
  if (i + 8 > n) return;
  float4 a = *(const float4*)(in + i);
  float4 b = *(const float4*)(in + i + 4);
  v8bf o;
  o[0] = (bf16)a.x; o[1] = (bf16)a.y; o[2] = (bf16)a.z; o[3] = (bf16)a.w;
  o[4] = (bf16)b.x; o[5] = (bf16)b.y; o[6] = (bf16)b.z; o[7] = (bf16)b.w;
  *(v8bf*)(out + i) = o;
}

// fp32[rows][cols] -> optional bf16 row-major copy + bf16 transpose.
__global__ void cvt_transpose_kernel(const float* __restrict__ in,
                                     bf16* __restrict__ out_row,
                                     bf16* __restrict__ out_tr,
                                     int rows, int cols) {
  size_t i = (size_t)blockIdx.x * blockDim.x + threadIdx.x;
  if (i >= (size_t)rows * cols) return;
  int r = (int)(i / cols), c = (int)(i % cols);
  float v = in[i];
  if (out_row) out_row[i] = (bf16)v;
  out_tr[(size_t)c * rows + r] = (bf16)v;
}

extern "C" void kernel_launch(void* const* d_in, const int* in_sizes, int n_in,
                              void* d_out, int out_size, void* d_ws, size_t ws_size,
                              hipStream_t stream) {
  (void)in_sizes; (void)n_in; (void)out_size; (void)ws_size;
  const float* x   = (const float*)d_in[0];
  const float* adj = (const float*)d_in[1];
  const float* W1  = (const float*)d_in[2];
  const float* b1  = (const float*)d_in[3];
  const float* W2  = (const float*)d_in[4];
  const float* b2  = (const float*)d_in[5];
  const float* W3  = (const float*)d_in[6];
  const float* b3  = (const float*)d_in[7];
  const float* W4  = (const float*)d_in[8];
  const float* b4  = (const float*)d_in[9];
  float* out = (float*)d_out;

  char* ws = (char*)d_ws;
  size_t off = 0;
  auto carve = [&](size_t bytes) -> char* {
    char* p = ws + off;
    off += (bytes + 255) & ~(size_t)255;
    return p;
  };
  bf16* adjb = (bf16*)carve((size_t)GN * GN * 2);        // 128 MB, L2-resident
  bf16* hA   = (bf16*)carve((size_t)GN * 256 * 2);       // h (row major)
  bf16* hAT  = (bf16*)carve((size_t)GN * 256 * 2);       // h^T
  bf16* hB   = (bf16*)carve((size_t)GN * 128 * 2);
  bf16* hBT  = (bf16*)carve((size_t)GN * 128 * 2);
  bf16* S    = (bf16*)carve((size_t)GN * 256 * 2);       // neighbor support
  bf16* W1t  = (bf16*)carve((size_t)512 * 128 * 2);
  bf16* W2t  = (bf16*)carve((size_t)256 * 128 * 2);
  bf16* W3t  = (bf16*)carve((size_t)256 * 128 * 2);
  bf16* W4t  = (bf16*)carve((size_t)256 * 64 * 2);

  // --- precision/layout prep ---
  cvt_f32_bf16_kernel<<<(unsigned)((size_t)GN * GN / 8 / 256), 256, 0, stream>>>(
      adj, adjb, (size_t)GN * GN);
  cvt_transpose_kernel<<<(unsigned)((size_t)GN * GNFEAT / 256), 256, 0, stream>>>(
      x, hA, hAT, GN, GNFEAT);
  cvt_transpose_kernel<<<(512 * 128) / 256, 256, 0, stream>>>(W1, nullptr, W1t, 512, 128);
  cvt_transpose_kernel<<<(256 * 128) / 256, 256, 0, stream>>>(W2, nullptr, W2t, 256, 128);
  cvt_transpose_kernel<<<(256 * 128) / 256, 256, 0, stream>>>(W3, nullptr, W3t, 256, 128);
  cvt_transpose_kernel<<<(256 * 64)  / 256, 256, 0, stream>>>(W4, nullptr, W4t, 256, 64);

  const dim3 blk(256);

  // --- layer 1: S = adj @ x  (N=256), then hB = relu([x|S] W1 + b1) ---
  gemm_bf16_kernel<128, false, 0><<<dim3(GN / 128, 2), blk, 0, stream>>>(
      adjb, nullptr, GN, 0, hAT, GN, GN, GN, 256, nullptr, S, nullptr, nullptr);
  gemm_bf16_kernel<128, true, 1><<<dim3(GN / 128, 1), blk, 0, stream>>>(
      hA, S, 0, 256, W1t, 512, 512, GN, 128, b1, hB, hBT, nullptr);

  // --- layer 2 ---
  gemm_bf16_kernel<128, false, 0><<<dim3(GN / 128, 1), blk, 0, stream>>>(
      adjb, nullptr, GN, 0, hBT, GN, GN, GN, 128, nullptr, S, nullptr, nullptr);
  gemm_bf16_kernel<128, true, 1><<<dim3(GN / 128, 1), blk, 0, stream>>>(
      hB, S, 0, 128, W2t, 256, 256, GN, 128, b2, hA, hAT, nullptr);

  // --- layer 3 ---
  gemm_bf16_kernel<128, false, 0><<<dim3(GN / 128, 1), blk, 0, stream>>>(
      adjb, nullptr, GN, 0, hAT, GN, GN, GN, 128, nullptr, S, nullptr, nullptr);
  gemm_bf16_kernel<128, true, 1><<<dim3(GN / 128, 1), blk, 0, stream>>>(
      hA, S, 0, 128, W3t, 256, 256, GN, 128, b3, hB, hBT, nullptr);

  // --- layer 4 (no relu, fp32 output) ---
  gemm_bf16_kernel<128, false, 0><<<dim3(GN / 128, 1), blk, 0, stream>>>(
      adjb, nullptr, GN, 0, hBT, GN, GN, GN, 128, nullptr, S, nullptr, nullptr);
  gemm_bf16_kernel<64, true, 2><<<dim3(GN / 128, 1), blk, 0, stream>>>(
      hB, S, 0, 128, W4t, 256, 256, GN, GNCLS, b4, nullptr, nullptr, out);
}
